// SwinTransformerBlock_87273735454885
// MI455X (gfx1250) — compile-verified, matrix operand order checked by reading, not run on hardware
//
#include <hip/hip_runtime.h>
#include <hip/hip_bf16.h>

typedef unsigned short u16;
typedef unsigned int   u32;
typedef __attribute__((ext_vector_type(16))) __bf16 v16bf;
typedef __attribute__((ext_vector_type(8)))  float  v8f;

#define HSEQ 4096
#define DIMC 512
#define NBATCH 16
#define MROWS (NBATCH * HSEQ)   // 65536 tokens

// ---------- bf16 helpers (bit-level, RNE) ----------
__device__ __forceinline__ u16 f2bf(float f) {
    u32 u = __float_as_uint(f);
    u32 r = u + 0x7fffu + ((u >> 16) & 1u);
    return (u16)(r >> 16);
}
__device__ __forceinline__ float bflo(u32 u) { return __uint_as_float(u << 16); }
__device__ __forceinline__ float bfhi(u32 u) { return __uint_as_float(u & 0xffff0000u); }
__device__ __forceinline__ u32 pack2(float a, float b) {
    return (u32)f2bf(a) | ((u32)f2bf(b) << 16);
}
__device__ __forceinline__ float dotu(u32 a, u32 b) {
    return bflo(a) * bflo(b) + bfhi(a) * bfhi(b);
}

// ---------- CDNA5 async global->LDS copy (ASYNCcnt-tracked, no VGPR staging) ----
// Generic pointers to LDS hold the LDS byte offset in their low 32 bits
// (ISA 10.2), so we can feed the VDST (LDS address)
// operand directly from a truncated generic pointer.
__device__ __forceinline__ void async_ld_b128(void* lds, const void* gptr) {
    u32 ldsaddr = (u32)(size_t)lds;
    asm volatile("global_load_async_to_lds_b128 %0, %1, off"
                 :: "v"(ldsaddr), "v"(gptr) : "memory");
}
__device__ __forceinline__ void wait_async0() {
    asm volatile("s_wait_asynccnt 0x0" ::: "memory");
}

// ---------- fp32 -> bf16 weight pre-conversion (once per launch) ----------
__global__ __launch_bounds__(256) void w_to_bf16(
    const float* __restrict__ w, u16* __restrict__ o) {
    int t = blockIdx.x * 256 + threadIdx.x;     // one thread per 8 floats
    const float4* p = (const float4*)w;
    float4 a = p[2 * t], b = p[2 * t + 1];
    uint4 v;
    v.x = pack2(a.x, a.y); v.y = pack2(a.z, a.w);
    v.z = pack2(b.x, b.y); v.w = pack2(b.z, b.w);
    ((uint4*)o)[t] = v;
}

// ---------- LayerNorm (+optional roll along seq) -> bf16 ----------
__global__ __launch_bounds__(256) void ln_roll_bf16(
    const float* __restrict__ x, const float* __restrict__ g,
    const float* __restrict__ bta, u16* __restrict__ out, int roll) {
    int wave = threadIdx.x >> 5, lane = threadIdx.x & 31;
    int grow = blockIdx.x * 8 + wave;            // destination row (shifted coords)
    int b = grow >> 12, j = grow & (HSEQ - 1);
    const float4* src = (const float4*)(x + ((size_t)b * HSEQ + ((j + roll) & (HSEQ - 1))) * DIMC);
    float4 v[4];
    float s = 0.f, q = 0.f;
    for (int i = 0; i < 4; ++i) {
        v[i] = src[lane + 32 * i];
        s += v[i].x + v[i].y + v[i].z + v[i].w;
        q += v[i].x * v[i].x + v[i].y * v[i].y + v[i].z * v[i].z + v[i].w * v[i].w;
    }
    for (int off = 16; off; off >>= 1) { s += __shfl_xor(s, off); q += __shfl_xor(q, off); }
    float mean = s * (1.f / DIMC);
    float var  = q * (1.f / DIMC) - mean * mean;
    float rs   = rsqrtf(var + 1e-5f);
    const float4* g4 = (const float4*)g;
    const float4* b4 = (const float4*)bta;
    uint2* dst = (uint2*)(out + (size_t)grow * DIMC);
    for (int i = 0; i < 4; ++i) {
        float4 gg = g4[lane + 32 * i], bb = b4[lane + 32 * i];
        float4 n;
        n.x = (v[i].x - mean) * rs * gg.x + bb.x;
        n.y = (v[i].y - mean) * rs * gg.y + bb.y;
        n.z = (v[i].z - mean) * rs * gg.z + bb.z;
        n.w = (v[i].w - mean) * rs * gg.w + bb.w;
        uint2 o2; o2.x = pack2(n.x, n.y); o2.y = pack2(n.z, n.w);
        dst[lane + 32 * i] = o2;
    }
}

// ---------- WMMA bf16 GEMM:  out = A(bf16, MxK) @ Bw(bf16, NxK)^T + bias ------
// Block tile 128(M) x 128(N) x 32(K); 256 threads = 8 waves (2 along M x 4
// along N), each wave owns a 64x32 patch = 4x2 v_wmma_f32_16x16x32_bf16 tiles
// (8 WMMA per 12 b128 fragment loads). Both tiles stream in via
// global_load_async_to_lds_b128 (ASYNCcnt), double-buffered; LDS rows padded
// to 80B (5 x b128) for conflict-free b128 fragment reads.
// A-fragment per ISA 16-bit A 16x32 layout: lane(half=L/16, idx=L%16) reads
// K chunks [8*half, +8) and [16+8*half, +8)  == two b128 loads.
// MODE 0: store bf16; MODE 1: exact-GELU then bf16; MODE 2: f32 = acc+bias+res.
union FragU { uint4 q[2]; v16bf v; };

template <int MODE>
__global__ __launch_bounds__(256) void gemm_wmma_bf16(
    const u16* __restrict__ A, const u16* __restrict__ Bw,
    const float* __restrict__ bias, u16* __restrict__ outB,
    float* __restrict__ outF, const float* __restrict__ res,
    int N, int K) {
    __shared__ uint4 ldsA[2][128][5];
    __shared__ uint4 ldsB[2][128][5];
    const int tid = threadIdx.x;
    const int bm = blockIdx.y * 128, bn = blockIdx.x * 128;
    const int lane = tid & 31, wv = tid >> 5;
    const int waveM = wv >> 2, waveN = wv & 3;       // 2 x 4 waves
    const int half = lane >> 4, idx = lane & 15;

    // each thread issues 2 A-chunks + 2 B-chunks per K-step (512 b128 per tile)
    auto issueTile = [&](int kt, int s) {
        for (int i = 0; i < 2; ++i) {
            int c = tid + 256 * i, row = c >> 2, seg = c & 3;
            async_ld_b128(&ldsA[s][row][seg],
                          A  + (size_t)(bm + row) * K + kt * 32 + seg * 8);
            async_ld_b128(&ldsB[s][row][seg],
                          Bw + (size_t)(bn + row) * K + kt * 32 + seg * 8);
        }
    };

    v8f acc[4][2];
    for (int mi = 0; mi < 4; ++mi)
        for (int ni = 0; ni < 2; ++ni)
            for (int e = 0; e < 8; ++e) acc[mi][ni][e] = 0.f;

    const int NT = K >> 5;
    issueTile(0, 0);
    for (int kt = 0; kt < NT; ++kt) {
        int s = kt & 1;
        wait_async0();                 // this thread's buf-s chunks have landed
        __syncthreads();               // everyone's landed; buf s^1 reads done
        if (kt + 1 < NT) issueTile(kt + 1, s ^ 1);   // overlap with WMMA below
        FragU a[4], b[2];
        for (int mi = 0; mi < 4; ++mi) {
            const uint4* p = &ldsA[s][waveM * 64 + mi * 16 + idx][0];
            a[mi].q[0] = p[half]; a[mi].q[1] = p[2 + half];
        }
        for (int ni = 0; ni < 2; ++ni) {
            const uint4* p = &ldsB[s][waveN * 32 + ni * 16 + idx][0];
            b[ni].q[0] = p[half]; b[ni].q[1] = p[2 + half];
        }
        for (int mi = 0; mi < 4; ++mi)
            for (int ni = 0; ni < 2; ++ni)
                acc[mi][ni] = __builtin_amdgcn_wmma_f32_16x16x32_bf16(
                    false, a[mi].v, false, b[ni].v, (short)0, acc[mi][ni], false, false);
    }

    // epilogue: D tile layout -> row = base + r + 8*half, col = base + idx
    for (int mi = 0; mi < 4; ++mi) {
        for (int ni = 0; ni < 2; ++ni) {
            int colg = bn + waveN * 32 + ni * 16 + idx;
            float bv = bias[colg];
            int rowb = bm + waveM * 64 + mi * 16 + 8 * half;
            for (int r = 0; r < 8; ++r) {
                size_t off = (size_t)(rowb + r) * N + colg;
                float val = acc[mi][ni][r] + bv;
                if (MODE == 0) {
                    outB[off] = f2bf(val);
                } else if (MODE == 1) {
                    val = 0.5f * val * (1.f + erff(val * 0.70710678118654752f));
                    outB[off] = f2bf(val);
                } else {
                    outF[off] = val + res[off];
                }
            }
        }
    }
}

// ---------- windowed attention ----------
// window w = b*512 + s1*64 + n0 ; element p -> shifted-seq j = n0*64 + p*8 + s1.
// Mask reduces to: n0==63 isolates token p==7 (-100 both ways).
// One thread per (window, head, q-row); output scattered to rolled(+4) seq order.
__global__ __launch_bounds__(256) void attn_win(
    const u16* __restrict__ qkv, u16* __restrict__ outp) {
    int t = blockIdx.x * 256 + threadIdx.x;        // 524288 threads
    int p = t & 7, h = (t >> 3) & 7, w = t >> 6;
    int b = w >> 9, rr = w & 511, s1 = rr >> 6, n0 = rr & 63;
    int jq = n0 * 64 + p * 8 + s1;
    const uint4* q4 = (const uint4*)(qkv + ((size_t)b * HSEQ + jq) * 1536 + h * 64);
    uint4 q[8];
    for (int c = 0; c < 8; ++c) q[c] = q4[c];

    float s[8];
    for (int kk = 0; kk < 8; ++kk) {
        size_t rowk = (size_t)b * HSEQ + n0 * 64 + kk * 8 + s1;
        const uint4* k4 = (const uint4*)(qkv + rowk * 1536 + 512 + h * 64);
        float acc = 0.f;
        for (int c = 0; c < 8; ++c) {
            uint4 kv = k4[c];
            acc += dotu(q[c].x, kv.x) + dotu(q[c].y, kv.y) +
                   dotu(q[c].z, kv.z) + dotu(q[c].w, kv.w);
        }
        s[kk] = acc * 0.125f;                      // HD^-0.5 = 1/8
    }
    if (n0 == 63) {
        bool mp = (p == 7);
        for (int kk = 0; kk < 8; ++kk)
            if (mp != (kk == 7)) s[kk] -= 100.f;
    }
    float mx = s[0];
    for (int kk = 1; kk < 8; ++kk) mx = fmaxf(mx, s[kk]);
    float sum = 0.f;
    for (int kk = 0; kk < 8; ++kk) { s[kk] = __expf(s[kk] - mx); sum += s[kk]; }
    float inv = 1.f / sum;

    float o_[64];
    for (int d = 0; d < 64; ++d) o_[d] = 0.f;
    for (int kk = 0; kk < 8; ++kk) {
        float pk = s[kk] * inv;
        size_t rowv = (size_t)b * HSEQ + n0 * 64 + kk * 8 + s1;
        const uint4* v4 = (const uint4*)(qkv + rowv * 1536 + 1024 + h * 64);
        for (int c = 0; c < 8; ++c) {
            uint4 vv = v4[c];
            o_[c * 8 + 0] += pk * bflo(vv.x); o_[c * 8 + 1] += pk * bfhi(vv.x);
            o_[c * 8 + 2] += pk * bflo(vv.y); o_[c * 8 + 3] += pk * bfhi(vv.y);
            o_[c * 8 + 4] += pk * bflo(vv.z); o_[c * 8 + 5] += pk * bfhi(vv.z);
            o_[c * 8 + 6] += pk * bflo(vv.w); o_[c * 8 + 7] += pk * bfhi(vv.w);
        }
    }
    int jf = (jq + 4) & (HSEQ - 1);                // window_reverse + roll(+4)
    uint4* od = (uint4*)(outp + ((size_t)b * HSEQ + jf) * DIMC + h * 64);
    for (int c = 0; c < 8; ++c) {
        uint4 ov;
        ov.x = pack2(o_[c * 8 + 0], o_[c * 8 + 1]);
        ov.y = pack2(o_[c * 8 + 2], o_[c * 8 + 3]);
        ov.z = pack2(o_[c * 8 + 4], o_[c * 8 + 5]);
        ov.w = pack2(o_[c * 8 + 6], o_[c * 8 + 7]);
        od[c] = ov;
    }
}

extern "C" void kernel_launch(void* const* d_in, const int* in_sizes, int n_in,
                              void* d_out, int out_size, void* d_ws, size_t ws_size,
                              hipStream_t stream) {
    const float* x      = (const float*)d_in[0];
    const float* n1g    = (const float*)d_in[1];
    const float* n1b    = (const float*)d_in[2];
    const float* qkv_w  = (const float*)d_in[3];
    const float* qkv_b  = (const float*)d_in[4];
    const float* proj_w = (const float*)d_in[5];
    const float* proj_b = (const float*)d_in[6];
    const float* n2g    = (const float*)d_in[7];
    const float* n2b    = (const float*)d_in[8];
    const float* fc1_w  = (const float*)d_in[9];
    const float* fc1_b  = (const float*)d_in[10];
    const float* fc2_w  = (const float*)d_in[11];
    const float* fc2_b  = (const float*)d_in[12];
    float* out = (float*)d_out;

    // workspace layout (peak ~326 MB, regions reused once consumed)
    char* ws = (char*)d_ws;
    u16* xs   = (u16*)(ws);                          //  64 MB: LN1(+roll) bf16
    u16* qkvb = (u16*)(ws + ((size_t)64  << 20));    // 192 MB: qkv bf16
    u16* ob   = (u16*)(ws + ((size_t)256 << 20));    //  64 MB: attn out bf16
    u16* hn   = (u16*)(ws);                          //  reuse: LN2 out bf16
    u16* f1   = (u16*)(ws + ((size_t)64  << 20));    // 256 MB reuse: fc1 out bf16
    char* wb  = ws + ((size_t)320 << 20);            //  ~6 MB: bf16 weights
    u16* qkv_wb  = (u16*)(wb);                       // 1536*512
    u16* proj_wb = (u16*)(wb + 0x180000);            // 512*512
    u16* fc1_wb  = (u16*)(wb + 0x200000);            // 2048*512
    u16* fc2_wb  = (u16*)(wb + 0x400000);            // 512*2048

    // 0) one-time (per launch) fp32 -> bf16 weight conversion
    w_to_bf16<<<(1536 * 512) / 2048, 256, 0, stream>>>(qkv_w,  qkv_wb);
    w_to_bf16<<<( 512 * 512) / 2048, 256, 0, stream>>>(proj_w, proj_wb);
    w_to_bf16<<<(2048 * 512) / 2048, 256, 0, stream>>>(fc1_w,  fc1_wb);
    w_to_bf16<<<( 512 * 2048) / 2048, 256, 0, stream>>>(fc2_w,  fc2_wb);

    // 1) LN1 + roll(-4)
    ln_roll_bf16<<<MROWS / 8, 256, 0, stream>>>(x, n1g, n1b, xs, 4);
    // 2) QKV GEMM (65536 x 1536 x 512)
    gemm_wmma_bf16<0><<<dim3(1536 / 128, MROWS / 128), 256, 0, stream>>>(
        xs, qkv_wb, qkv_b, qkvb, nullptr, nullptr, 1536, 512);
    // 3) windowed attention, scattered back to rolled sequence order
    attn_win<<<(MROWS * 8) / 256, 256, 0, stream>>>(qkvb, ob);
    // 4) proj GEMM + residual(x) -> x1 (fp32) in d_out
    gemm_wmma_bf16<2><<<dim3(512 / 128, MROWS / 128), 256, 0, stream>>>(
        ob, proj_wb, proj_b, nullptr, out, x, 512, 512);
    // 5) LN2
    ln_roll_bf16<<<MROWS / 8, 256, 0, stream>>>(out, n2g, n2b, hn, 0);
    // 6) FC1 GEMM + exact GELU (65536 x 2048 x 512)
    gemm_wmma_bf16<1><<<dim3(2048 / 128, MROWS / 128), 256, 0, stream>>>(
        hn, fc1_wb, fc1_b, f1, nullptr, nullptr, 2048, 512);
    // 7) FC2 GEMM + residual(x1) -> d_out (65536 x 512 x 2048)
    gemm_wmma_bf16<2><<<dim3(512 / 128, MROWS / 128), 256, 0, stream>>>(
        f1, fc2_wb, fc2_b, nullptr, out, out, 512, 2048);
}
